// LGNN_79087527789189
// MI455X (gfx1250) — matrix operand order repaired and checked by reading, not compiled, not run on hardware
//
#include <hip/hip_runtime.h>
#include <math.h>

// ---------------- problem constants (match reference) ----------------
#define NN     5000
#define EE     50000
#define ELG    200000
#define DIM    512
#define HEADS  8
#define DK     64
#define LAYERS 2

// ---------------- types for WMMA ----------------
typedef float  f32x8   __attribute__((ext_vector_type(8)));
typedef __bf16 bf16x16 __attribute__((ext_vector_type(16)));
typedef __bf16 bf16x8  __attribute__((ext_vector_type(8)));
typedef __bf16 bf16x4  __attribute__((ext_vector_type(4)));

union FragU { bf16x16 v; bf16x8 h[2]; };

// =====================================================================
// WMMA GEMM:  O[M x Nn] = epilogue( A[M x K] @ W[K x Nn] )
// epilogue: + bias[n] (optional), + R[m,n] residual (optional), relu (opt)
// Tile 128x128, BK=32, 256 threads = 8 waves; wave = 32x64 (2x4 WMMA).
// K multiple of 32, Nn multiple of 128 (true here: 512 or 2048).
// LDS k-stride padded to 40 elems (80B): keeps 16B alignment of the
// 16/32/48B intra-row chunks while spreading rows across banks.
// Interior blocks (m0+BM<=M) take a branch-free staging/epilogue path.
// =====================================================================
#define BM 128
#define BN 128
#define BK 32
#define KP 40

__global__ __launch_bounds__(256)
void wmma_gemm_kernel(const float* __restrict__ A, const float* __restrict__ W,
                      const float* __restrict__ bias, const float* __restrict__ R,
                      float* __restrict__ O, int M, int K, int Nn, int doRelu)
{
    __shared__ __align__(16) __bf16 As[BM][KP];   // [m][k]
    __shared__ __align__(16) __bf16 Bs[BN][KP];   // [n][k]  (W transposed)

    const int tid  = threadIdx.x;
    const int lane = tid & 31;
    const int wave = tid >> 5;          // 0..7
    const int wm   = (wave >> 1) * 32;  // wave row offset inside tile
    const int wn   = (wave & 1)  * 64;  // wave col offset inside tile
    const int m0   = blockIdx.y * BM;
    const int n0   = blockIdx.x * BN;
    const bool fullTile = (m0 + BM) <= M;   // wave-uniform

    const int lm = lane & 15;           // row (A) / col (B,C) within 16
    const int kg = lane >> 4;           // k-group select

    // W-stage mapping: each thread owns a 4k x 4n block of the 32x128 tile
    const int sn4 = (tid & 31) * 4;     // n within tile: 0..124
    const int sk4 = (tid >> 5) * 4;     // k within tile: 0..28

    // A-stage mapping: thread covers 4 float4 slots (row i>>3, col (i&7)*4)
    const int arow = tid >> 3;          // base row for r=0 (0..31), +32 per r
    const int ac4  = (tid & 7) * 4;

    f32x8 acc[2][4] = {};

    for (int kk = 0; kk < K; kk += BK) {
        // ---- A tile 128x32: f32 -> bf16 row-major, b64 stores ----
        if (fullTile) {
            #pragma unroll
            for (int r = 0; r < 4; ++r) {
                int row = arow + r * 32;
                float4 va = *(const float4*)(A + (size_t)(m0 + row) * K + kk + ac4);
                bf16x4 p;
                p[0] = (__bf16)va.x; p[1] = (__bf16)va.y;
                p[2] = (__bf16)va.z; p[3] = (__bf16)va.w;
                *(bf16x4*)&As[row][ac4] = p;
            }
        } else {
            #pragma unroll
            for (int r = 0; r < 4; ++r) {
                int row = arow + r * 32;
                float4 va = make_float4(0.f, 0.f, 0.f, 0.f);
                if (m0 + row < M)
                    va = *(const float4*)(A + (size_t)(m0 + row) * K + kk + ac4);
                bf16x4 p;
                p[0] = (__bf16)va.x; p[1] = (__bf16)va.y;
                p[2] = (__bf16)va.z; p[3] = (__bf16)va.w;
                *(bf16x4*)&As[row][ac4] = p;
            }
        }
        // ---- W tile 32x128: 4 coalesced b128 row loads, register-transpose,
        //      4x ds_store_b64 into Bs[n][k] ----
        {
            const float* wp = W + (size_t)(kk + sk4) * Nn + n0 + sn4;
            float4 r0 = *(const float4*)(wp);
            float4 r1 = *(const float4*)(wp + Nn);
            float4 r2 = *(const float4*)(wp + 2 * (size_t)Nn);
            float4 r3 = *(const float4*)(wp + 3 * (size_t)Nn);
            bf16x4 c;
            c[0]=(__bf16)r0.x; c[1]=(__bf16)r1.x; c[2]=(__bf16)r2.x; c[3]=(__bf16)r3.x;
            *(bf16x4*)&Bs[sn4 + 0][sk4] = c;
            c[0]=(__bf16)r0.y; c[1]=(__bf16)r1.y; c[2]=(__bf16)r2.y; c[3]=(__bf16)r3.y;
            *(bf16x4*)&Bs[sn4 + 1][sk4] = c;
            c[0]=(__bf16)r0.z; c[1]=(__bf16)r1.z; c[2]=(__bf16)r2.z; c[3]=(__bf16)r3.z;
            *(bf16x4*)&Bs[sn4 + 2][sk4] = c;
            c[0]=(__bf16)r0.w; c[1]=(__bf16)r1.w; c[2]=(__bf16)r2.w; c[3]=(__bf16)r3.w;
            *(bf16x4*)&Bs[sn4 + 3][sk4] = c;
        }
        __syncthreads();

        // ---- fragments per ISA 7.12.2 layouts ----
        FragU a[2], b[4];
        #pragma unroll
        for (int t = 0; t < 2; ++t) {
            int mr = wm + t * 16 + lm;        // A: lane holds row (lane&15)
            a[t].h[0] = *(const bf16x8*)&As[mr][kg * 8];        // k = 8*kg+0..7
            a[t].h[1] = *(const bf16x8*)&As[mr][16 + kg * 8];   // k = 16+8*kg+0..7
        }
        #pragma unroll
        for (int t = 0; t < 4; ++t) {
            int nc = wn + t * 16 + lm;        // B: lane holds col (lane&15)
            b[t].h[0] = *(const bf16x8*)&Bs[nc][kg * 8];        // k = 8*kg+0..7
            b[t].h[1] = *(const bf16x8*)&Bs[nc][16 + kg * 8];   // k = 16+8*kg+0..7
        }
        #pragma unroll
        for (int ti = 0; ti < 2; ++ti)
            #pragma unroll
            for (int tj = 0; tj < 4; ++tj)
                acc[ti][tj] = __builtin_amdgcn_wmma_f32_16x16x32_bf16(
                    false, a[ti].v, false, b[tj].v,
                    (short)0, acc[ti][tj], false, false);
        __syncthreads();
    }

    // ---- epilogue: C/D layout -> global ----
    #pragma unroll
    for (int ti = 0; ti < 2; ++ti) {
        #pragma unroll
        for (int tj = 0; tj < 4; ++tj) {
            int n = n0 + wn + tj * 16 + lm;
            float bv = bias ? bias[n] : 0.0f;
            int mb = m0 + wm + ti * 16 + (lane >> 4) * 8;
            if (fullTile) {
                #pragma unroll
                for (int r = 0; r < 8; ++r) {
                    size_t idx = (size_t)(mb + r) * Nn + n;
                    float o = acc[ti][tj][r] + bv;
                    if (R) o += R[idx];
                    if (doRelu) o = fmaxf(o, 0.0f);
                    O[idx] = o;
                }
            } else {
                #pragma unroll
                for (int r = 0; r < 8; ++r) {
                    if (mb + r < M) {
                        size_t idx = (size_t)(mb + r) * Nn + n;
                        float o = acc[ti][tj][r] + bv;
                        if (R) o += R[idx];
                        if (doRelu) o = fmaxf(o, 0.0f);
                        O[idx] = o;
                    }
                }
            }
        }
    }
}

// =====================================================================
// attention: one wave per (edge, head).  score = exp(clip(k.q/8,±10))
// wv[dst] += (v[src](+e))*score ; z[dst] += score   (f32 atomics)
// =====================================================================
__global__ __launch_bounds__(256)
void attn_kernel(const float* __restrict__ q, const float* __restrict__ k,
                 const float* __restrict__ v, const float* __restrict__ ef,
                 const int* __restrict__ src, const int* __restrict__ dst,
                 float* __restrict__ wv, float* __restrict__ z, int Ne)
{
    int wid  = blockIdx.x * (blockDim.x >> 5) + (threadIdx.x >> 5);
    int lane = threadIdx.x & 31;
    if (wid >= Ne * HEADS) return;        // wave-uniform exit
    int e = wid >> 3;                     // / HEADS(=8)
    int h = wid & 7;
    int s = src[e], d = dst[e];

    const float2* kb = (const float2*)(k + ((size_t)s * HEADS + h) * DK);
    const float2* qb = (const float2*)(q + ((size_t)d * HEADS + h) * DK);
    const float2* vb = (const float2*)(v + ((size_t)s * HEADS + h) * DK);
    float2 kv = kb[lane], vv = vb[lane], qv = qb[lane];
    if (ef) {
        const float2* eb = (const float2*)(ef + ((size_t)e * HEADS + h) * DK);
        float2 ev = eb[lane];
        kv.x += ev.x; kv.y += ev.y;
        vv.x += ev.x; vv.y += ev.y;
    }
    float p = kv.x * qv.x + kv.y * qv.y;
    #pragma unroll
    for (int o = 16; o > 0; o >>= 1) p += __shfl_xor(p, o, 32);
    float t = fminf(fmaxf(p * 0.125f, -10.0f), 10.0f);   // /sqrt(64), clip
    float sc = expf(t);

    float* wb = wv + ((size_t)d * HEADS + h) * DK;
    atomicAdd(&wb[2 * lane],     vv.x * sc);
    atomicAdd(&wb[2 * lane + 1], vv.y * sc);
    if (lane == 0) atomicAdd(&z[(size_t)d * HEADS + h], sc);
}

__global__ __launch_bounds__(256)
void divz_kernel(float* __restrict__ wv, const float* __restrict__ z, size_t total)
{
    size_t i = (size_t)blockIdx.x * blockDim.x + threadIdx.x;
    if (i < total) wv[i] = wv[i] / z[i >> 6];   // i>>6 == seg*HEADS + h
}

// =====================================================================
// LayerNorm in place: one wave per row of 512 (single pass over memory)
// =====================================================================
__global__ __launch_bounds__(256)
void ln_kernel(float* __restrict__ x, const float* __restrict__ g,
               const float* __restrict__ b, int M)
{
    int wid  = blockIdx.x * (blockDim.x >> 5) + (threadIdx.x >> 5);
    int lane = threadIdx.x & 31;
    if (wid >= M) return;
    float* row = x + (size_t)wid * DIM;

    float vals[DIM / 32];
    float s = 0.0f;
    #pragma unroll
    for (int c = 0; c < DIM / 32; ++c) {
        vals[c] = row[lane + c * 32];
        s += vals[c];
    }
    #pragma unroll
    for (int o = 16; o > 0; o >>= 1) s += __shfl_xor(s, o, 32);
    float mean = s * (1.0f / DIM);

    float vs = 0.0f;
    #pragma unroll
    for (int c = 0; c < DIM / 32; ++c) {
        float dlt = vals[c] - mean;
        vals[c] = dlt; vs += dlt * dlt;
    }
    #pragma unroll
    for (int o = 16; o > 0; o >>= 1) vs += __shfl_xor(vs, o, 32);
    float inv = rsqrtf(vs * (1.0f / DIM) + 1e-5f);
    #pragma unroll
    for (int c = 0; c < DIM / 32; ++c) {
        int col = lane + c * 32;
        row[col] = vals[c] * inv * g[col] + b[col];
    }
}

// lg_x[e,:] = rel_embed[edge_feat[e],:]
__global__ __launch_bounds__(256)
void gather_rel_kernel(float* __restrict__ out, const float* __restrict__ rel,
                       const int* __restrict__ feat, size_t total)
{
    size_t i = (size_t)blockIdx.x * blockDim.x + threadIdx.x;
    if (i >= total) return;
    size_t e = i >> 9, c = i & 511;
    out[i] = rel[(size_t)feat[e] * DIM + c];
}

// out[e,:] += x[idx[e],:]
__global__ __launch_bounds__(256)
void add_gather_kernel(float* __restrict__ out, const float* __restrict__ x,
                       const int* __restrict__ idx, size_t total)
{
    size_t i = (size_t)blockIdx.x * blockDim.x + threadIdx.x;
    if (i >= total) return;
    size_t e = i >> 9, c = i & 511;
    out[i] += x[(size_t)idx[e] * DIM + c];
}

// ---------------- host-side helpers ----------------
static void gemm(const float* A, const float* W, const float* bias, const float* R,
                 float* O, int M, int K, int Nn, int relu, hipStream_t s)
{
    dim3 grid(Nn / BN, (M + BM - 1) / BM);
    hipLaunchKernelGGL(wmma_gemm_kernel, grid, dim3(256), 0, s, A, W, bias, R, O, M, K, Nn, relu);
}

static inline int blocks_for(size_t total, int bs) { return (int)((total + bs - 1) / bs); }

// weight slot order inside each of the n_/e_ groups
enum { WQ = 0, BQ, WK, WV, WO, BO, LNG, LNB, F1, FB1, F2, FB2, FLNG, FLNB };

extern "C" void kernel_launch(void* const* d_in, const int* in_sizes, int n_in,
                              void* d_out, int out_size, void* d_ws, size_t ws_size,
                              hipStream_t stream)
{
    (void)in_sizes; (void)n_in; (void)out_size; (void)ws_size;

    const float* x_in  = (const float*)d_in[0];
    const int*   efeat = (const int*)d_in[1];
    const int*   src   = (const int*)d_in[2];
    const int*   dst   = (const int*)d_in[3];
    const int*   lgsrc = (const int*)d_in[4];
    const int*   lgdst = (const int*)d_in[5];
    const float* rel   = (const float*)d_in[6];
    const float* pn[14], *pe[14];
    for (int j = 0; j < 14; ++j) pn[j] = (const float*)d_in[7 + j];
    for (int j = 0; j < 14; ++j) pe[j] = (const float*)d_in[21 + j];

    const size_t N512  = (size_t)NN * DIM;
    const size_t E512  = (size_t)EE * DIM;
    const size_t N2048 = (size_t)NN * 4 * DIM;
    const size_t E2048 = (size_t)EE * 4 * DIM;

    // x and lg_x live in d_out:  out = (x, lg_x) concatenated
    float* x  = (float*)d_out;
    float* lg = (float*)d_out + N512;

    // -------- workspace layout (floats) --------
    float* base = (float*)d_ws;
    size_t off = 0;
    auto alloc = [&](size_t n) { float* p = base + off; off += (n + 255) & ~(size_t)255; return p; };
    float* q_n    = alloc(N512);
    float* k_n    = alloc(N512);
    float* v_n    = alloc(N512);
    float* wv_n   = alloc(N512);
    float* z_n    = alloc((size_t)NN * HEADS);
    float* mid_n  = alloc(N2048);
    float* xa_n   = alloc(N512);
    float* outx_n = alloc(N512);
    float* ebig   = alloc(E2048);          // overlays: qe/ke/ve then FFN mid
    float* qe     = ebig;
    float* ke     = ebig + E512;
    float* ve     = ebig + 2 * E512;
    float* mid_e  = ebig;                  // reused after attention
    float* wv_e   = alloc(E512);
    float* z_e    = alloc((size_t)EE * HEADS);
    float* ea_e   = alloc(E512);
    float* outlg  = alloc(E512);

    // -------- init: x <- input, lg_x <- rel_embed[edge_feat] --------
    hipMemcpyAsync(x, x_in, N512 * sizeof(float), hipMemcpyDeviceToDevice, stream);
    hipLaunchKernelGGL(gather_rel_kernel, dim3(blocks_for(E512, 256)), dim3(256), 0, stream,
                       lg, rel, efeat, E512);

    const size_t W2 = (size_t)DIM * DIM;          // 512*512
    const size_t WF = (size_t)DIM * 4 * DIM;      // 512*2048

    for (int i = 0; i < LAYERS; ++i) {
        // ================= node update =================
        gemm(x, pn[WQ] + i * W2, pn[BQ] + i * DIM, nullptr, q_n, NN, DIM, DIM, 0, stream);
        gemm(x, pn[WK] + i * W2, nullptr,          nullptr, k_n, NN, DIM, DIM, 0, stream);
        gemm(x, pn[WV] + i * W2, nullptr,          nullptr, v_n, NN, DIM, DIM, 0, stream);

        hipMemsetAsync(wv_n, 0, N512 * sizeof(float), stream);
        hipMemsetAsync(z_n,  0, (size_t)NN * HEADS * sizeof(float), stream);
        hipLaunchKernelGGL(attn_kernel, dim3(blocks_for((size_t)EE * HEADS * 32, 256)), dim3(256),
                           0, stream, q_n, k_n, v_n, lg, src, dst, wv_n, z_n, EE);
        hipLaunchKernelGGL(divz_kernel, dim3(blocks_for(N512, 256)), dim3(256), 0, stream,
                           wv_n, z_n, N512);

        gemm(wv_n, pn[WO] + i * W2, pn[BO] + i * DIM, x, xa_n, NN, DIM, DIM, 0, stream);
        hipLaunchKernelGGL(ln_kernel, dim3(blocks_for((size_t)NN * 32, 256)), dim3(256), 0, stream,
                           xa_n, pn[LNG] + i * DIM, pn[LNB] + i * DIM, NN);
        gemm(xa_n, pn[F1] + i * WF, pn[FB1] + i * 4 * DIM, nullptr, mid_n, NN, DIM, 4 * DIM, 1, stream);
        gemm(mid_n, pn[F2] + i * WF, pn[FB2] + i * DIM, xa_n, outx_n, NN, 4 * DIM, DIM, 0, stream);
        hipLaunchKernelGGL(ln_kernel, dim3(blocks_for((size_t)NN * 32, 256)), dim3(256), 0, stream,
                           outx_n, pn[FLNG] + i * DIM, pn[FLNB] + i * DIM, NN);

        // ================= edge update (uses PRE-update x) =================
        gemm(lg, pe[WQ] + i * W2, pe[BQ] + i * DIM, nullptr, qe, EE, DIM, DIM, 0, stream);
        hipLaunchKernelGGL(add_gather_kernel, dim3(blocks_for(E512, 256)), dim3(256), 0, stream,
                           qe, x, dst, E512);
        gemm(lg, pe[WK] + i * W2, nullptr, nullptr, ke, EE, DIM, DIM, 0, stream);
        hipLaunchKernelGGL(add_gather_kernel, dim3(blocks_for(E512, 256)), dim3(256), 0, stream,
                           ke, x, src, E512);
        gemm(lg, pe[WV] + i * W2, nullptr, nullptr, ve, EE, DIM, DIM, 0, stream);
        hipLaunchKernelGGL(add_gather_kernel, dim3(blocks_for(E512, 256)), dim3(256), 0, stream,
                           ve, x, src, E512);

        hipMemsetAsync(wv_e, 0, E512 * sizeof(float), stream);
        hipMemsetAsync(z_e,  0, (size_t)EE * HEADS * sizeof(float), stream);
        hipLaunchKernelGGL(attn_kernel, dim3(blocks_for((size_t)ELG * HEADS * 32, 256)), dim3(256),
                           0, stream, qe, ke, ve, nullptr, lgsrc, lgdst, wv_e, z_e, ELG);
        hipLaunchKernelGGL(divz_kernel, dim3(blocks_for(E512, 256)), dim3(256), 0, stream,
                           wv_e, z_e, E512);

        gemm(wv_e, pe[WO] + i * W2, pe[BO] + i * DIM, lg, ea_e, EE, DIM, DIM, 0, stream);
        hipLaunchKernelGGL(ln_kernel, dim3(blocks_for((size_t)EE * 32, 256)), dim3(256), 0, stream,
                           ea_e, pe[LNG] + i * DIM, pe[LNB] + i * DIM, EE);
        gemm(ea_e, pe[F1] + i * WF, pe[FB1] + i * 4 * DIM, nullptr, mid_e, EE, DIM, 4 * DIM, 1, stream);
        gemm(mid_e, pe[F2] + i * WF, pe[FB2] + i * DIM, ea_e, outlg, EE, 4 * DIM, DIM, 0, stream);
        hipLaunchKernelGGL(ln_kernel, dim3(blocks_for((size_t)EE * 32, 256)), dim3(256), 0, stream,
                           outlg, pe[FLNG] + i * DIM, pe[FLNB] + i * DIM, EE);

        // ================= commit =================
        hipMemcpyAsync(x,  outx_n, N512 * sizeof(float), hipMemcpyDeviceToDevice, stream);
        hipMemcpyAsync(lg, outlg,  E512 * sizeof(float), hipMemcpyDeviceToDevice, stream);
    }
}
